// DNN_KalmanNet_arch2_62277025792489
// MI455X (gfx1250) — compile-verified, matrix operand order checked by reading, not run on hardware
//
#include <hip/hip_runtime.h>

typedef __attribute__((ext_vector_type(16))) _Float16 v16h;
typedef __attribute__((ext_vector_type(8)))  float    v8f;
typedef __attribute__((ext_vector_type(4)))  float    f32x4;
typedef __attribute__((ext_vector_type(4)))  unsigned u32x4;
typedef __attribute__((ext_vector_type(2)))  unsigned u32x2;

#define RS      880      // LDS row stride in f16 (multiple of 8 -> 16B-aligned rows)
#define NWAVES  2
#define TROWS   16

// per-wave LDS column map (all offsets multiples of 8; K ranges padded to x32,
// padding guaranteed zero by the initial LDS clear)
#define BUF1 0           // 128: FC5-out / in_S / in_FC4
#define BUFA 128         // 160: in_Sigma = [out_Q | FC6-out] (+16 pad)
#define BUFG 288         // 128: GRU gate scratch [r | z]
#define BUFS 416         // 128: [out_Sigma | out_S | K_vec] (+16 pad)
#define BUF2 544         // 320: FC2a hidden (overlays staged inputs, written late)
// staged inputs (dead before FC2a writes BUF2):
#define CX5  544
#define CX6  576
#define COBS 608
#define CHQ  640
#define CHS  704
#define CHSS 768

// packed-weight fragment offsets (1 fragment = 1KB = 64 u32x4)
#define FO_W5    0
#define FO_GQIH  5
#define FO_GQHH  41
#define FO_W6    65
#define FO_GGIH  70
#define FO_GGHH  130
#define FO_W7    154
#define FO_W1    159
#define FO_GSIH  161
#define FO_GSHH  170
#define FO_W2A   173
#define FO_W2B   233
#define FO_W3    253
#define FO_W4    261
#define NFRAGS   277

union AF  { v16h h; u32x4 q[2]; };
union H4  { _Float16 h[4]; u32x2 u; };

__device__ __forceinline__ float sigmoidf_(float x) {
  return 1.0f / (1.0f + __expf(-x));
}

__device__ __forceinline__ v8f wmma16(v16h a, v16h b, v8f c) {
  return __builtin_amdgcn_wmma_f32_16x16x32_f16(false, a, false, b, (short)0, c,
                                                false, false);
}

// ---- A fragment (16 rows x 32 K) from LDS, branchless: 2x ds_load_b128 ----
__device__ __forceinline__ v16h load_a_lds(const _Float16* base, int colOff,
                                           int kOff) {
  int lane = threadIdx.x & 31, m = lane & 15, g = lane >> 4;
  const _Float16* rp = base + m * RS + colOff + kOff + 8 * g;
  AF af;
  af.q[0] = *(const u32x4*)(rp);
  af.q[1] = *(const u32x4*)(rp + 16);
  return af.h;
}

// ---- B fragment from pre-packed f16 workspace: 2x global_load_b128 --------
__device__ __forceinline__ v16h load_b_packed(const u32x4* __restrict__ wp,
                                              int fragIdx) {
  int lane = threadIdx.x & 31;
  const u32x4* p = wp + (size_t)fragIdx * 64 + lane * 2;
  AF af;
  af.q[0] = p[0];
  af.q[1] = p[1];
  return af.h;
}

// ---- B fragment fallback from W[O,I] f32 row-major, branchless ------------
template <int I, int O>
__device__ __forceinline__ v16h load_b_g(const float* __restrict__ W, int kOff,
                                         int nOff) {
  int lane = threadIdx.x & 31, n = lane & 15, g = lane >> 4;
  const float* rp = W + (size_t)(nOff + n) * I;
  int KA = kOff + 8 * g;
  int KB = KA + 16;
  KA = (KA + 8 <= I) ? KA : 0;  // padded groups: A columns are exact zeros
  KB = (KB + 8 <= I) ? KB : 0;
  f32x4 p0 = *(const f32x4*)(rp + KA);
  f32x4 p1 = *(const f32x4*)(rp + KA + 4);
  f32x4 p2 = *(const f32x4*)(rp + KB);
  f32x4 p3 = *(const f32x4*)(rp + KB + 4);
  v16h b;
#pragma unroll
  for (int i = 0; i < 4; ++i) {
    b[i]      = (_Float16)p0[i];
    b[4 + i]  = (_Float16)p1[i];
    b[8 + i]  = (_Float16)p2[i];
    b[12 + i] = (_Float16)p3[i];
  }
  return b;
}

// ---- D fragment stores -----------------------------------------------------
template <bool RELU>
__device__ __forceinline__ void store_d_lds(const v8f& d, _Float16* base,
                                            int colOff) {
  int lane = threadIdx.x & 31, n = lane & 15, g = lane >> 4;
  _Float16* cp = base + colOff + n;
#pragma unroll
  for (int r = 0; r < 8; ++r) {
    float x = d[r];
    if (RELU) x = x > 0.0f ? x : 0.0f;
    cp[(r + 8 * g) * RS] = (_Float16)x;
  }
}

template <bool RELU>
__device__ __forceinline__ void store_d_g(const v8f& d, float* __restrict__ out,
                                          int ldo, int row0, int nOff) {
  int lane = threadIdx.x & 31, n = lane & 15, g = lane >> 4;
#pragma unroll
  for (int r = 0; r < 8; ++r) {
    float x = d[r];
    if (RELU) x = x > 0.0f ? x : 0.0f;
    out[(size_t)(row0 + r + 8 * g) * ldo + nOff + n] = x;
  }
}

// ---- stage 16 rows x F f32 from global into LDS f16 (coalesced b128) ------
template <int F>
__device__ __forceinline__ void stage16(const float* __restrict__ src, int row0,
                                        _Float16* base, int col) {
  int lane = threadIdx.x & 31;
  constexpr int C4 = F >> 2;
  constexpr int NC = TROWS * C4;
  for (int i = lane; i < NC; i += 32) {
    int row = i / C4;
    int c4 = (i - row * C4) << 2;
    f32x4 v = *(const f32x4*)(src + (size_t)(row0 + row) * F + c4);
    H4 h;
    h.h[0] = (_Float16)v[0];
    h.h[1] = (_Float16)v[1];
    h.h[2] = (_Float16)v[2];
    h.h[3] = (_Float16)v[3];
    *(u32x2*)(base + row * RS + col + c4) = h.u;
  }
}

// ---- Generic FC layer: out[16,O] = act(x @ W.T + b) ------------------------
template <int IN, int O, bool RELU, bool PACKED, bool TOLDS, bool TOGLOB>
__device__ __forceinline__ void fc_layer(const _Float16* abase, int aCol,
                                         const float* __restrict__ W,
                                         const float* __restrict__ b,
                                         const u32x4* __restrict__ wsp,
                                         int fragOff, _Float16* base,
                                         int outCol, float* __restrict__ gout,
                                         int gld, int row0) {
  constexpr int KT = (IN + 31) / 32;
  int n = threadIdx.x & 15;
#pragma unroll 1
  for (int no = 0; no < O; no += 16) {
    float bias = b[no + n];  // hoisted: overlaps with the GEMM below
    v8f acc = {};
#pragma unroll
    for (int kt = 0; kt < KT; ++kt) {
      v16h a = load_a_lds(abase, aCol, kt * 32);
      v16h bb = PACKED ? load_b_packed(wsp, fragOff + (no >> 4) * KT + kt)
                       : load_b_g<IN, O>(W, kt * 32, no);
      acc = wmma16(a, bb, acc);
    }
#pragma unroll
    for (int r = 0; r < 8; ++r) acc[r] += bias;
    if (TOLDS) store_d_lds<RELU>(acc, base, outCol + no);
    if (TOGLOB) store_d_g<RELU>(acc, gout, gld, row0, no);
  }
}

// ---- GRU cell (single step, PyTorch gate math); x and h both in LDS --------
template <int IN, int H, bool PACKED, bool TOGLOB>
__device__ __forceinline__ void gru_layer(
    _Float16* base, int xCol, int hCol, const float* __restrict__ w_ih,
    const float* __restrict__ w_hh, const float* __restrict__ b_ih,
    const float* __restrict__ b_hh, const u32x4* __restrict__ wsp, int foIH,
    int foHH, int outCol, float* __restrict__ gout, int row0) {
  constexpr int KTX = (IN + 31) / 32;
  constexpr int KTH = (H + 31) / 32;
  int lane = threadIdx.x & 31, n = lane & 15, g = lane >> 4;
#pragma unroll 1
  for (int no = 0; no < 3 * H; no += 16) {
    float bi = b_ih[no + n];
    float bh = b_hh[no + n];
    v8f gi = {};
#pragma unroll
    for (int kt = 0; kt < KTX; ++kt) {
      v16h a = load_a_lds(base, xCol, kt * 32);
      v16h bb = PACKED ? load_b_packed(wsp, foIH + (no >> 4) * KTX + kt)
                       : load_b_g<IN, 3 * H>(w_ih, kt * 32, no);
      gi = wmma16(a, bb, gi);
    }
    v8f gh = {};
#pragma unroll
    for (int kt = 0; kt < KTH; ++kt) {
      v16h a = load_a_lds(base, hCol, kt * 32);
      v16h bb = PACKED ? load_b_packed(wsp, foHH + (no >> 4) * KTH + kt)
                       : load_b_g<H, 3 * H>(w_hh, kt * 32, no);
      gh = wmma16(a, bb, gh);
    }
    if (no < 2 * H) {
#pragma unroll
      for (int r = 0; r < 8; ++r) {
        float s = sigmoidf_(gi[r] + gh[r] + bi + bh);
        base[(r + 8 * g) * RS + BUFG + no + n] = (_Float16)s;
      }
    } else {
      int c0 = no - 2 * H;
#pragma unroll
      for (int r = 0; r < 8; ++r) {
        int row = r + 8 * g;
        const _Float16* rb = base + row * RS;
        float rv = (float)rb[BUFG + c0 + n];
        float zv = (float)rb[BUFG + H + c0 + n];
        float c = tanhf(gi[r] + bi + rv * (gh[r] + bh));
        float hv = (float)rb[hCol + c0 + n];
        float hn = (1.0f - zv) * c + zv * hv;
        base[row * RS + outCol + c0 + n] = (_Float16)hn;
        if (TOGLOB) gout[(size_t)(row0 + row) * H + c0 + n] = hn;
      }
    }
  }
}

struct KParams {
  const float *obs, *inn, *evol, *upd, *hQ, *hSig, *hS;
  const float *gq_wih, *gq_whh, *gq_bih, *gq_bhh;
  const float *gg_wih, *gg_whh, *gg_bih, *gg_bhh;
  const float *gs_wih, *gs_whh, *gs_bih, *gs_bhh;
  const float *W1, *b1, *W2a, *b2a, *W2b, *b2b, *W3, *b3, *W4, *b4;
  const float *W5, *b5, *W6, *b6, *W7, *b7;
  float *out_K, *out_hQ, *out_FC4, *out_S;
};

// ---- one-time weight pre-pack: f32 [O,I] -> f16 WMMA B-fragments in d_ws ---
struct PackParams { const float* W[14]; };

__global__ __launch_bounds__(256) void pack_weights(PackParams PP,
                                                    u32x4* __restrict__ ws) {
  const int I_[14]  = {8, 80, 64, 8, 144, 64, 8, 64, 96, 16, 80, 320, 48, 128};
  const int OF_[14] = {FO_W5, FO_GQIH, FO_GQHH, FO_W6, FO_GGIH, FO_GGHH,
                       FO_W7, FO_W1,   FO_GSIH, FO_GSHH, FO_W2A, FO_W2B,
                       FO_W3, FO_W4};
  int t = blockIdx.x * blockDim.x + threadIdx.x;
  int frag = t >> 5, lane = t & 31;
  if (frag >= NFRAGS) return;
  int L = 0;
  while (L + 1 < 14 && frag >= OF_[L + 1]) ++L;
  const float* W = PP.W[L];
  int I = I_[L];
  int KT = (I + 31) >> 5;
  int f = frag - OF_[L];
  int no = (f / KT) << 4;
  int kOff = (f % KT) << 5;
  int n = lane & 15, g = lane >> 4;
  const float* rp = W + (size_t)(no + n) * I;
  AF af;
#pragma unroll
  for (int v = 0; v < 8; ++v) {
    int K = ((v < 4) ? 2 * v : 16 + 2 * (v - 4)) + 8 * g + kOff;
    float x0 = (K < I) ? rp[K] : 0.0f;
    float x1 = (K + 1 < I) ? rp[K + 1] : 0.0f;
    af.h[2 * v] = (_Float16)x0;
    af.h[2 * v + 1] = (_Float16)x1;
  }
  u32x4* dst = ws + (size_t)frag * 64 + lane * 2;
  dst[0] = af.q[0];
  dst[1] = af.q[1];
}

template <bool PACKED>
__global__ __launch_bounds__(NWAVES * 32) void knet_wmma_kernel(
    KParams P, const u32x4* __restrict__ wsp, int Btot) {
  __shared__ __align__(16) _Float16 smem[NWAVES * TROWS * RS];
  int wave = threadIdx.x >> 5;
  int row0 = (blockIdx.x * NWAVES + wave) * TROWS;
  if (row0 + TROWS > Btot) return;
  _Float16* base = smem + wave * TROWS * RS;
  int lane = threadIdx.x & 31;

  // zero this wave's whole LDS region (16*RS*2 = 28160 B = 55 * 512B)
  {
    u32x4 z = {0u, 0u, 0u, 0u};
    char* p = (char*)base;
#pragma unroll 1
    for (int ofs = lane * 16; ofs < TROWS * RS * 2; ofs += 32 * 16)
      *(u32x4*)(p + ofs) = z;
  }

  // stage all global f32 inputs into LDS f16 (coalesced b128 loads)
  stage16<8>(P.evol, row0, base, CX5);
  stage16<8>(P.upd, row0, base, CX6);
  stage16<4>(P.obs, row0, base, COBS);
  stage16<4>(P.inn, row0, base, COBS + 4);
  stage16<64>(P.hQ, row0, base, CHQ);
  stage16<64>(P.hSig, row0, base, CHS);
  stage16<16>(P.hS, row0, base, CHSS);

  // FC5 -> BUF1[0:80]
  fc_layer<8, 80, true, PACKED, true, false>(base, CX5, P.W5, P.b5, wsp, FO_W5,
                                             base, BUF1, nullptr, 0, row0);
  // GRU_Q -> BUFA[0:64] + global h_new_Q
  gru_layer<80, 64, PACKED, true>(base, BUF1, CHQ, P.gq_wih, P.gq_whh,
                                  P.gq_bih, P.gq_bhh, wsp, FO_GQIH, FO_GQHH,
                                  BUFA, P.out_hQ, row0);
  // FC6 -> BUFA[64:144]
  fc_layer<8, 80, true, PACKED, true, false>(base, CX6, P.W6, P.b6, wsp, FO_W6,
                                             base, BUFA + 64, nullptr, 0, row0);
  // GRU_Sigma -> BUFS[0:64]
  gru_layer<144, 64, PACKED, false>(base, BUFA, CHS, P.gg_wih, P.gg_whh,
                                    P.gg_bih, P.gg_bhh, wsp, FO_GGIH, FO_GGHH,
                                    BUFS, nullptr, row0);
  // FC7 -> BUF1[16:96]
  fc_layer<8, 80, true, PACKED, true, false>(base, COBS, P.W7, P.b7, wsp, FO_W7,
                                             base, BUF1 + 16, nullptr, 0, row0);
  // FC1 -> BUF1[0:16]
  fc_layer<64, 16, true, PACKED, true, false>(base, BUFS, P.W1, P.b1, wsp,
                                              FO_W1, base, BUF1, nullptr, 0,
                                              row0);
  // GRU_S -> BUFS[64:80] + global out_S
  gru_layer<96, 16, PACKED, true>(base, BUF1, CHSS, P.gs_wih, P.gs_whh,
                                  P.gs_bih, P.gs_bhh, wsp, FO_GSIH, FO_GSHH,
                                  BUFS + 64, P.out_S, row0);
  // FC2a -> BUF2[0:320]
  fc_layer<80, 320, true, PACKED, true, false>(base, BUFS, P.W2a, P.b2a, wsp,
                                               FO_W2A, base, BUF2, nullptr, 0,
                                               row0);
  // FC2b -> K_vec: BUFS[80:112] + global
  fc_layer<320, 32, false, PACKED, true, true>(base, BUF2, P.W2b, P.b2b, wsp,
                                               FO_W2B, base, BUFS + 80, P.out_K,
                                               32, row0);
  // copy out_Sigma (BUFS[0:64]) -> BUF1[0:64] for contiguous in_FC4
  for (int idx = lane; idx < TROWS * 16; idx += 32) {
    int row = idx >> 4, cw = idx & 15;
    *(u32x2*)(base + row * RS + BUF1 + 4 * cw) =
        *(const u32x2*)(base + row * RS + BUFS + 4 * cw);
  }
  // FC3 -> BUF1[64:128]
  fc_layer<48, 64, true, PACKED, true, false>(base, BUFS + 64, P.W3, P.b3, wsp,
                                              FO_W3, base, BUF1 + 64, nullptr,
                                              0, row0);
  // FC4 -> global out_FC4
  fc_layer<128, 64, true, PACKED, false, true>(base, BUF1, P.W4, P.b4, wsp,
                                               FO_W4, nullptr, 0, P.out_FC4, 64,
                                               row0);
}

extern "C" void kernel_launch(void* const* d_in, const int* in_sizes, int n_in,
                              void* d_out, int out_size, void* d_ws,
                              size_t ws_size, hipStream_t stream) {
  (void)n_in; (void)out_size;
  const int B = in_sizes[0] / 4;  // obs_diff is [B,4]
  KParams P;
  P.obs = (const float*)d_in[0];
  P.inn = (const float*)d_in[1];
  P.evol = (const float*)d_in[2];
  P.upd = (const float*)d_in[3];
  P.hQ = (const float*)d_in[4];
  P.hSig = (const float*)d_in[5];
  P.hS = (const float*)d_in[6];
  P.gq_wih = (const float*)d_in[7];
  P.gq_whh = (const float*)d_in[8];
  P.gq_bih = (const float*)d_in[9];
  P.gq_bhh = (const float*)d_in[10];
  P.gg_wih = (const float*)d_in[11];
  P.gg_whh = (const float*)d_in[12];
  P.gg_bih = (const float*)d_in[13];
  P.gg_bhh = (const float*)d_in[14];
  P.gs_wih = (const float*)d_in[15];
  P.gs_whh = (const float*)d_in[16];
  P.gs_bih = (const float*)d_in[17];
  P.gs_bhh = (const float*)d_in[18];
  P.W1 = (const float*)d_in[19];
  P.b1 = (const float*)d_in[20];
  P.W2a = (const float*)d_in[21];
  P.b2a = (const float*)d_in[22];
  P.W2b = (const float*)d_in[23];
  P.b2b = (const float*)d_in[24];
  P.W3 = (const float*)d_in[25];
  P.b3 = (const float*)d_in[26];
  P.W4 = (const float*)d_in[27];
  P.b4 = (const float*)d_in[28];
  P.W5 = (const float*)d_in[29];
  P.b5 = (const float*)d_in[30];
  P.W6 = (const float*)d_in[31];
  P.b6 = (const float*)d_in[32];
  P.W7 = (const float*)d_in[33];
  P.b7 = (const float*)d_in[34];

  float* out = (float*)d_out;
  P.out_K = out;                      // [B,32]
  P.out_hQ = out + (size_t)B * 32;    // [B,64]
  P.out_FC4 = out + (size_t)B * 96;   // [B,64]
  P.out_S = out + (size_t)B * 160;    // [B,16]

  int tiles = B / TROWS;
  int blocks = (tiles + NWAVES - 1) / NWAVES;

  bool packed = ws_size >= (size_t)NFRAGS * 1024;
  if (packed) {
    PackParams PK;
    PK.W[0] = P.W5;    PK.W[1] = P.gq_wih; PK.W[2] = P.gq_whh;
    PK.W[3] = P.W6;    PK.W[4] = P.gg_wih; PK.W[5] = P.gg_whh;
    PK.W[6] = P.W7;    PK.W[7] = P.W1;     PK.W[8] = P.gs_wih;
    PK.W[9] = P.gs_whh; PK.W[10] = P.W2a;  PK.W[11] = P.W2b;
    PK.W[12] = P.W3;   PK.W[13] = P.W4;
    pack_weights<<<(NFRAGS * 32 + 255) / 256, 256, 0, stream>>>(
        PK, (u32x4*)d_ws);
    knet_wmma_kernel<true><<<blocks, NWAVES * 32, 0, stream>>>(
        P, (const u32x4*)d_ws, B);
  } else {
    knet_wmma_kernel<false><<<blocks, NWAVES * 32, 0, stream>>>(P, nullptr, B);
  }
}
